// multihead_self_attention_14370960572841
// MI455X (gfx1250) — matrix operand order repaired and checked by reading, not compiled, not run on hardware
//
#include <hip/hip_runtime.h>
#include <hip/hip_bf16.h>

// ---------------------------------------------------------------------------
// Types
// ---------------------------------------------------------------------------
typedef __attribute__((ext_vector_type(16))) __bf16 v16bf;
typedef __attribute__((ext_vector_type(4)))  __bf16 v4bf;
typedef __attribute__((ext_vector_type(8)))  float  v8f;
typedef __attribute__((ext_vector_type(4)))  float  v4f;
typedef __attribute__((ext_vector_type(4)))  int    i4;

#define B_   2
#define S_   2048
#define D_   1024
#define H_   16
#define DK_  64
#define NEG_INF (-3.0e38f)

// LDS row paddings (elements) to spread rows across banks while keeping
// 16-byte alignment for ds_load_b128 (stride in dwords coprime-ish with 64).
#define KPAD 72   // K tile rows: 64 + 8   (144 B, 36 dwords)
#define VPAD 40   // V tile rows: 32 + 8   ( 80 B, 20 dwords)
#define PPAD 40   // P tile rows: 32 + 8

// Async global->LDS path (ASYNCcnt), if the toolchain exposes the builtin.
// Probe result: param type is `int __attribute__((vector_size(16))) *`.
#if __has_builtin(__builtin_amdgcn_global_load_async_to_lds_b128)
#define HAVE_ASYNC_LDS 1
typedef int avi4 __attribute__((vector_size(16)));
#endif

__device__ __forceinline__ void wait_async_zero() {
#if defined(HAVE_ASYNC_LDS)
#if __has_builtin(__builtin_amdgcn_s_wait_asynccnt)
  __builtin_amdgcn_s_wait_asynccnt(0);
#else
  asm volatile("s_wait_asynccnt 0x0" ::: "memory");
#endif
#endif
}

union FragU { v16bf v; i4 d[2]; };

// 16 contiguous bf16 -> B fragment (contraction dim contiguous per lane)
__device__ __forceinline__ v16bf load_b16(const __bf16* p) {
  FragU u;
  u.d[0] = *(const i4*)(p);
  u.d[1] = *(const i4*)(p + 8);
  return u.v;
}
// A fragment: elements 0..15 = row[koff..koff+15]
__device__ __forceinline__ v16bf load_a(const __bf16* row, int koff) {
  FragU u;
  u.d[0] = *(const i4*)(row + koff);
  u.d[1] = *(const i4*)(row + koff + 8);
  return u.v;
}

__device__ __forceinline__ v8f wmma_bf16(v16bf a, v16bf b, v8f c) {
  return __builtin_amdgcn_wmma_f32_16x16x32_bf16(
      /*neg_a=*/false, a, /*neg_b=*/false, b,
      /*c_mod=*/(short)0, c, /*reuse_a=*/false, /*reuse_b=*/false);
}

// reductions over the 16-lane half-wave (rows of C-layout live per half)
__device__ __forceinline__ float redmax16(float v) {
  v = fmaxf(v, __shfl_xor(v, 1, 32));
  v = fmaxf(v, __shfl_xor(v, 2, 32));
  v = fmaxf(v, __shfl_xor(v, 4, 32));
  v = fmaxf(v, __shfl_xor(v, 8, 32));
  return v;
}
__device__ __forceinline__ float redsum16(float v) {
  v += __shfl_xor(v, 1, 32);
  v += __shfl_xor(v, 2, 32);
  v += __shfl_xor(v, 4, 32);
  v += __shfl_xor(v, 8, 32);
  return v;
}

// ---------------------------------------------------------------------------
// Kernel 1: fp32 -> bf16 cast (4 elems / thread)
// ---------------------------------------------------------------------------
__global__ void cast_bf16_kernel(const float* __restrict__ in,
                                 __bf16* __restrict__ out, int n4) {
  int i = blockIdx.x * blockDim.x + threadIdx.x;
  if (i >= n4) return;
  v4f f = *(const v4f*)(in + 4 * i);
  v4bf o;
  o[0] = (__bf16)f[0]; o[1] = (__bf16)f[1];
  o[2] = (__bf16)f[2]; o[3] = (__bf16)f[3];
  *(v4bf*)(out + 4 * i) = o;
}

// ---------------------------------------------------------------------------
// Kernel 2: QKV projection + RoPE, 64x32 register-blocked wave tiles.
//   out[m, o] = sum_d x[m,d] * Wqkv[o,d] ; M = B*S = 4096, N = 3*D = 3072
//   q,k -> [B,H,S,DK] bf16 (RoPE applied), v -> [B,H,DK,S] bf16 (transposed)
// ---------------------------------------------------------------------------
__global__ void qkv_rope_kernel(const __bf16* __restrict__ xb,
                                const __bf16* __restrict__ wqkv,
                                __bf16* __restrict__ qb,
                                __bf16* __restrict__ kbuf,
                                __bf16* __restrict__ vtb) {
  const int lane = threadIdx.x & 31;
  const int wave = threadIdx.x >> 5;
  const int tile = blockIdx.x * 8 + wave;      // 6144 wave tiles
  const int mt = tile / 96;                    // 64 tiles of 64 rows
  const int nt = tile % 96;                    // 96 tiles of 32 cols
  const int n    = lane & 15;
  const int half = lane >> 4;
  const int akoff = half * 16;
  const int bkoff = half * 16;

  const __bf16* arow[4];
  #pragma unroll
  for (int im = 0; im < 4; ++im)
    arow[im] = xb + (size_t)(mt * 64 + im * 16 + n) * D_;
  const __bf16* brow[2];
  #pragma unroll
  for (int jn = 0; jn < 2; ++jn)
    brow[jn] = wqkv + (size_t)(nt * 32 + jn * 16 + n) * D_ + bkoff;

  v8f acc[4][2] = {};
  for (int kb = 0; kb < D_; kb += 32) {
    v16bf a[4], b[2];
    #pragma unroll
    for (int im = 0; im < 4; ++im) {
      a[im] = load_a(arow[im] + kb, akoff);
      __builtin_prefetch(arow[im] + kb + 64, 0, 1);
    }
    #pragma unroll
    for (int jn = 0; jn < 2; ++jn) {
      b[jn] = load_b16(brow[jn] + kb);
      __builtin_prefetch(brow[jn] + kb + 64, 0, 1);
    }
    #pragma unroll
    for (int im = 0; im < 4; ++im)
      #pragma unroll
      for (int jn = 0; jn < 2; ++jn)
        acc[im][jn] = wmma_bf16(a[im], b[jn], acc[im][jn]);
  }

  #pragma unroll
  for (int jn = 0; jn < 2; ++jn) {
    const int o     = nt * 32 + jn * 16 + n;
    const int which = o >> 10;            // 0=q 1=k 2=v (uniform across wave)
    const int h     = (o >> 6) & 15;
    const int dk    = o & 63;
    // inv_freq = theta^(-(dk&~1)/64); ln(10000) = 9.2103403720
    const float inv = __expf(-(float)(dk & ~1) * (9.2103403720f / 64.0f));
    #pragma unroll
    for (int im = 0; im < 4; ++im) {
      #pragma unroll
      for (int r = 0; r < 8; ++r) {
        const int row  = mt * 64 + im * 16 + r + 8 * half;  // token row
        const int bidx = row >> 11;
        const int s    = row & (S_ - 1);
        float val = acc[im][jn][r];
        if (which == 2) {
          // V transposed: [B,H,DK,S]
          vtb[((size_t)(bidx * H_ + h) * DK_ + dk) * S_ + s] = (__bf16)val;
        } else {
          float part = __shfl_xor(val, 1, 32);   // rotation partner
          float sn, cs;
          __sincosf((float)s * inv, &sn, &cs);
          float res = (lane & 1) ? (val * cs + part * sn)
                                 : (val * cs - part * sn);
          size_t idx = ((size_t)(bidx * H_ + h) * S_ + s) * DK_ + dk;
          if (which == 0) qb[idx]   = (__bf16)res;
          else            kbuf[idx] = (__bf16)res;
        }
      }
    }
  }
}

// ---------------------------------------------------------------------------
// Kernel 3: causal flash attention.
//   Block = 4 waves = 4 consecutive 32-query tiles of one (b,h).
//   K (32x64) and V^T (64x32) tiles staged cooperatively in LDS (async
//   global->LDS when available) and shared by all 4 waves; Q in registers;
//   P bounced via wave-private LDS.
// ---------------------------------------------------------------------------
__global__ void attn_kernel(const __bf16* __restrict__ qb,
                            const __bf16* __restrict__ kbuf,
                            const __bf16* __restrict__ vtb,
                            __bf16* __restrict__ attnb) {
  __shared__ __attribute__((aligned(16))) __bf16 ldsK[32 * KPAD];
  __shared__ __attribute__((aligned(16))) __bf16 ldsV[64 * VPAD];
  __shared__ __attribute__((aligned(16))) __bf16 ldsP[4][32 * PPAD];
  const int tid  = threadIdx.x;        // 0..127
  const int lane = tid & 31;
  const int wave = tid >> 5;
  const int bh = blockIdx.x >> 4;      // 32 (b,h) pairs
  const int qg = blockIdx.x & 15;      // 16 query groups of 128
  const int qt = qg * 4 + wave;        // this wave's 32-query tile
  const int n    = lane & 15;
  const int half = lane >> 4;
  const int akoff = half * 16;
  __bf16* P = ldsP[wave];

  // Q fragments in registers: [query subtile][d-half]
  v16bf aq[2][2];
  #pragma unroll
  for (int im = 0; im < 2; ++im) {
    const __bf16* qrow =
        qb + ((size_t)(bh * S_ + qt * 32 + im * 16 + n)) * DK_;
    aq[im][0] = load_a(qrow, akoff);
    aq[im][1] = load_a(qrow + 32, akoff);
  }

  v8f oacc[2][4] = {};
  float m_r[2][8], l_r[2][8];
  #pragma unroll
  for (int im = 0; im < 2; ++im)
    #pragma unroll
    for (int r = 0; r < 8; ++r) { m_r[im][r] = NEG_INF; l_r[im][r] = 0.0f; }

  const int nkeys_w   = qt * 32 + 32;        // this wave's causal extent
  const int nkeys_max = qg * 128 + 128;      // block's max extent (uniform)

  for (int kb = 0; kb < nkeys_max; kb += 32) {
    // ---- cooperative stage: K tile 32x64, V^T tile 64x32 (256 chunks each)
    #pragma unroll
    for (int c = tid; c < 256; c += 128) {
      const int kr = c >> 3, kc = (c & 7) * 8;     // K: 8 chunks per row
      const __bf16* kg = &kbuf[((size_t)(bh * S_ + kb + kr)) * DK_ + kc];
      const int vr = c >> 2, vc = (c & 3) * 8;     // V: 4 chunks per row
      const __bf16* vg = &vtb[((size_t)(bh * DK_ + vr)) * S_ + kb + vc];
#if defined(HAVE_ASYNC_LDS)
      __builtin_amdgcn_global_load_async_to_lds_b128(
          (avi4*)kg, (avi4*)&ldsK[kr * KPAD + kc], 0, 0);
      __builtin_amdgcn_global_load_async_to_lds_b128(
          (avi4*)vg, (avi4*)&ldsV[vr * VPAD + vc], 0, 0);
#else
      *(i4*)&ldsK[kr * KPAD + kc] = *(const i4*)kg;
      *(i4*)&ldsV[vr * VPAD + vc] = *(const i4*)vg;
#endif
      __builtin_prefetch(kg + 32 * DK_, 0, 1);
      __builtin_prefetch(vg + 32, 0, 1);
    }
    wait_async_zero();   // drain this wave's ASYNCcnt before the barrier
    __syncthreads();

    if (kb < nkeys_w) {   // wave-uniform predicate
      // ---- K B-fragments from LDS: [key half][d half] --------------------
      v16bf bd[2][2];
      #pragma unroll
      for (int t = 0; t < 2; ++t) {
        const __bf16* krow = &ldsK[(t * 16 + n) * KPAD + half * 16];
        bd[t][0] = load_b16(krow);
        bd[t][1] = load_b16(krow + 32);
      }
      // ---- per query subtile: scores, mask, online softmax, P -> LDS -----
      #pragma unroll
      for (int im = 0; im < 2; ++im) {
        v8f s0 = {}, s1 = {};
        s0 = wmma_bf16(aq[im][1], bd[0][1], s0);
        s0 = wmma_bf16(aq[im][0], bd[0][0], s0);
        s1 = wmma_bf16(aq[im][1], bd[1][1], s1);
        s1 = wmma_bf16(aq[im][0], bd[1][0], s1);
        #pragma unroll
        for (int r = 0; r < 8; ++r) {
          const int qi = qt * 32 + im * 16 + r + 8 * half;
          float a0 = s0[r] * 0.125f;
          float a1 = s1[r] * 0.125f;
          if (kb + n      > qi) a0 = NEG_INF;
          if (kb + 16 + n > qi) a1 = NEG_INF;
          const float tmax = redmax16(fmaxf(a0, a1));
          const float mn   = fmaxf(m_r[im][r], tmax);
          const float corr = __expf(m_r[im][r] - mn);
          m_r[im][r] = mn;
          const float p0 = __expf(a0 - mn);
          const float p1 = __expf(a1 - mn);
          l_r[im][r] = l_r[im][r] * corr + redsum16(p0 + p1);
          oacc[im][0][r] *= corr; oacc[im][1][r] *= corr;
          oacc[im][2][r] *= corr; oacc[im][3][r] *= corr;
          const int m = im * 16 + r + 8 * half;
          P[m * PPAD + n]      = (__bf16)p0;
          P[m * PPAD + 16 + n] = (__bf16)p1;
        }
      }
      __builtin_amdgcn_wave_barrier();   // order wave-private LDS st -> ld
      // ---- O += P * V (V fragment reused by both query subtiles) ---------
      v16bf ap[2];
      ap[0] = load_a(&P[n * PPAD], akoff);
      ap[1] = load_a(&P[(16 + n) * PPAD], akoff);
      #pragma unroll
      for (int j = 0; j < 4; ++j) {
        v16bf bv = load_b16(&ldsV[(j * 16 + n) * VPAD + half * 16]);
        oacc[0][j] = wmma_bf16(ap[0], bv, oacc[0][j]);
        oacc[1][j] = wmma_bf16(ap[1], bv, oacc[1][j]);
      }
      __builtin_amdgcn_wave_barrier();
    }
    __syncthreads();   // protect ldsK/ldsV before next stage
  }

  // ---- normalize and store attn output [B,S,D] bf16 ------------------------
  const int b = bh >> 4, h = bh & 15;
  #pragma unroll
  for (int im = 0; im < 2; ++im) {
    #pragma unroll
    for (int r = 0; r < 8; ++r) {
      const int s  = qt * 32 + im * 16 + r + 8 * half;
      const float rl = 1.0f / l_r[im][r];
      #pragma unroll
      for (int j = 0; j < 4; ++j) {
        attnb[((size_t)(b * S_ + s)) * D_ + h * DK_ + j * 16 + n] =
            (__bf16)(oacc[im][j][r] * rl);
      }
    }
  }
}

// ---------------------------------------------------------------------------
// Kernel 4: output projection, 64x32 register-blocked, fp32 result.
//   out[m, o] = sum_d attn[m,d] * Wo[o,d] ; M=4096, N=1024
// ---------------------------------------------------------------------------
__global__ void proj_kernel(const __bf16* __restrict__ attnb,
                            const __bf16* __restrict__ wob,
                            float* __restrict__ out) {
  const int lane = threadIdx.x & 31;
  const int wave = threadIdx.x >> 5;
  const int tile = blockIdx.x * 8 + wave;     // 2048 wave tiles
  const int mt = tile >> 5;                   // 64 tiles of 64 rows
  const int nt = tile & 31;                   // 32 tiles of 32 cols
  const int n    = lane & 15;
  const int half = lane >> 4;
  const int akoff = half * 16;
  const int bkoff = half * 16;

  const __bf16* arow[4];
  #pragma unroll
  for (int im = 0; im < 4; ++im)
    arow[im] = attnb + (size_t)(mt * 64 + im * 16 + n) * D_;
  const __bf16* brow[2];
  #pragma unroll
  for (int jn = 0; jn < 2; ++jn)
    brow[jn] = wob + (size_t)(nt * 32 + jn * 16 + n) * D_ + bkoff;

  v8f acc[4][2] = {};
  for (int kb = 0; kb < D_; kb += 32) {
    v16bf a[4], b[2];
    #pragma unroll
    for (int im = 0; im < 4; ++im) {
      a[im] = load_a(arow[im] + kb, akoff);
      __builtin_prefetch(arow[im] + kb + 64, 0, 1);
    }
    #pragma unroll
    for (int jn = 0; jn < 2; ++jn) {
      b[jn] = load_b16(brow[jn] + kb);
      __builtin_prefetch(brow[jn] + kb + 64, 0, 1);
    }
    #pragma unroll
    for (int im = 0; im < 4; ++im)
      #pragma unroll
      for (int jn = 0; jn < 2; ++jn)
        acc[im][jn] = wmma_bf16(a[im], b[jn], acc[im][jn]);
  }
  #pragma unroll
  for (int im = 0; im < 4; ++im)
    #pragma unroll
    for (int jn = 0; jn < 2; ++jn)
      #pragma unroll
      for (int r = 0; r < 8; ++r) {
        const int row = mt * 64 + im * 16 + r + 8 * half;
        out[(size_t)row * D_ + nt * 32 + jn * 16 + n] = acc[im][jn][r];
      }
}

// ---------------------------------------------------------------------------
// Host launcher
// ---------------------------------------------------------------------------
extern "C" void kernel_launch(void* const* d_in, const int* in_sizes, int n_in,
                              void* d_out, int out_size, void* d_ws, size_t ws_size,
                              hipStream_t stream) {
  const float* x    = (const float*)d_in[0];   // [2,2048,1024]
  const float* Wqkv = (const float*)d_in[1];   // [3072,1024]
  const float* Wo   = (const float*)d_in[2];   // [1024,1024]
  float* out = (float*)d_out;

  // workspace layout (bf16 buffers)
  char* base = (char*)d_ws;
  const size_t N_X    = (size_t)B_ * S_ * D_;       // 4,194,304
  const size_t N_WQKV = (size_t)3 * D_ * D_;        // 3,145,728
  const size_t N_WO   = (size_t)D_ * D_;            // 1,048,576
  __bf16* xb    = (__bf16*)(base);
  __bf16* wqkvb = (__bf16*)(base + 2 * N_X);
  __bf16* wob   = (__bf16*)(base + 2 * (N_X + N_WQKV));
  __bf16* qb    = (__bf16*)(base + 2 * (N_X + N_WQKV + N_WO));
  __bf16* kbuf  = (__bf16*)(base + 2 * (N_X + N_WQKV + N_WO + N_X));
  __bf16* vtb   = (__bf16*)(base + 2 * (N_X + N_WQKV + N_WO + 2 * N_X));
  __bf16* attnb = (__bf16*)(base + 2 * (N_X + N_WQKV + N_WO + 3 * N_X));
  const size_t needed = 2 * (N_X + N_WQKV + N_WO + 4 * N_X);
  if (ws_size < needed) return;

  // 1) casts
  cast_bf16_kernel<<<(int)((N_X / 4 + 255) / 256), 256, 0, stream>>>(x, xb, (int)(N_X / 4));
  cast_bf16_kernel<<<(int)((N_WQKV / 4 + 255) / 256), 256, 0, stream>>>(Wqkv, wqkvb, (int)(N_WQKV / 4));
  cast_bf16_kernel<<<(int)((N_WO / 4 + 255) / 256), 256, 0, stream>>>(Wo, wob, (int)(N_WO / 4));

  // 2) QKV projection + RoPE : 64*96 wave tiles (64x32), 8 waves/block
  qkv_rope_kernel<<<768, 256, 0, stream>>>(xb, wqkvb, qb, kbuf, vtb);

  // 3) flash attention : 32 bh * 16 query groups (128 q / block), 4 waves
  attn_kernel<<<512, 128, 0, stream>>>(qb, kbuf, vtb, attnb);

  // 4) output projection : 64*32 wave tiles (64x32), 8 waves/block
  proj_kernel<<<256, 256, 0, stream>>>(attnb, wob, out);
}
